// MixLlamaMLP_59279138619703
// MI455X (gfx1250) — matrix operand order repaired and checked by reading, not compiled, not run on hardware
//
#include <hip/hip_runtime.h>

// ---------------------------------------------------------------------------
// MixLlamaMLP for MI455X (gfx1250): bf16 WMMA GEMMs, fp32 accumulate,
// global->LDS staging via CDNA5 async DMA (global_load_async_to_lds_b128).
//   y = (silu(x Wg^T) * (x Wu^T)) Wd^T
// x:[4096,4096] Wg,Wu:[11008,4096] Wd:[4096,11008], all fp32 row-major
// ---------------------------------------------------------------------------

typedef __bf16 bf16;
typedef __attribute__((ext_vector_type(16))) __bf16 v16bf;
typedef __attribute__((ext_vector_type(8)))  __bf16 bf16x8;
typedef __attribute__((ext_vector_type(8)))  float  v8f;

#define D_MODEL 4096
#define D_FF    11008
#define N_TOK   4096

#define BM 128
#define BN 128
#define BK 32
#define LDS_STRIDE 40   // BK + 8 pad: 80B row pitch = 20 banks -> conflict-free frag reads
#define TILE_BYTES (BM * LDS_STRIDE * 2)   // 10240 B per tile per buffer

// CDNA5 async global->LDS DMA (ISA §15.18.3 op 98), GVS mode:
//   lds[vdst + ioffset] = mem[saddr + vaddr + ioffset]; tracked by ASYNCcnt.
__device__ __forceinline__ void async_ld_b128(unsigned lds_addr, unsigned voff,
                                              const void* sbase) {
  asm volatile("global_load_async_to_lds_b128 %0, %1, %2"
               :: "v"(lds_addr), "v"(voff), "s"(sbase) : "memory");
}
__device__ __forceinline__ void wait_async0() {
  asm volatile("s_wait_asynccnt 0" ::: "memory");
}
__device__ __forceinline__ unsigned lds_off(const void* p) {
  return (unsigned)(size_t)p;   // low 32 bits of LDS-aperture address = LDS byte offset
}

__device__ __forceinline__ bf16 f2bf(float f) {
  unsigned u = __builtin_bit_cast(unsigned, f);
  u += 0x7FFFu + ((u >> 16) & 1u);            // round-to-nearest-even
  unsigned short s = (unsigned short)(u >> 16);
  return __builtin_bit_cast(bf16, s);
}

__global__ void __launch_bounds__(256)
cvt_f32_bf16(const float* __restrict__ in, bf16* __restrict__ out, int n8) {
  int i = blockIdx.x * blockDim.x + threadIdx.x;
  int stride = gridDim.x * blockDim.x;
  for (; i < n8; i += stride) {
    const float4* p = (const float4*)(in + (size_t)i * 8);
    float4 a = p[0], b = p[1];
    bf16x8 o;
    o[0] = f2bf(a.x); o[1] = f2bf(a.y); o[2] = f2bf(a.z); o[3] = f2bf(a.w);
    o[4] = f2bf(b.x); o[5] = f2bf(b.y); o[6] = f2bf(b.z); o[7] = f2bf(b.w);
    *(bf16x8*)(out + (size_t)i * 8) = o;
  }
}

__device__ __forceinline__ v8f wmma_bf16(v16bf a, v16bf b, v8f c) {
  return __builtin_amdgcn_wmma_f32_16x16x32_bf16(
      /*neg_a=*/false, a, /*neg_b=*/false, b,
      /*c_mod=*/(short)0, c, /*reuse_a=*/false, /*reuse_b=*/false);
}

// A fragment (16x32 bf16, ISA 7.12.2): lanes 0-15 hold K 0..7 & 16..23 of row M=lane,
// lanes 16-31 hold K 8..15 & 24..31.  Two ds_load_b128 per fragment.
__device__ __forceinline__ v16bf load_frag_a(const bf16* t, int row, int lane) {
  int e = row * LDS_STRIDE + ((lane >> 4) << 3);
  bf16x8 lo = *(const bf16x8*)(t + e);
  bf16x8 hi = *(const bf16x8*)(t + e + 16);
  return __builtin_shufflevector(lo, hi, 0,1,2,3,4,5,6,7,8,9,10,11,12,13,14,15);
}
// B fragment (32x16 bf16): column N=lane%16; lanes 0-15 hold K 0..7 & 16..23, lanes 16-31 hold K 8..15 & 24..31.
__device__ __forceinline__ v16bf load_frag_b(const bf16* t, int row, int lane) {
  int e = row * LDS_STRIDE + ((lane >> 4) << 3);
  bf16x8 lo = *(const bf16x8*)(t + e);
  bf16x8 hi = *(const bf16x8*)(t + e + 16);
  return __builtin_shufflevector(lo, hi, 0,1,2,3,4,5,6,7,8,9,10,11,12,13,14,15);
}

// ---------------------------------------------------------------------------
// Kernel 1: C_g = X Wg^T, C_u = X Wu^T, H = silu(C_g)*C_u  (bf16 out)
// grid (D_FF/BN=86, N_TOK/BM=32), 256 threads = 8 waves (2 wave_m x 4 wave_n)
// ---------------------------------------------------------------------------
__global__ void __launch_bounds__(256)
gate_up_kernel(const bf16* __restrict__ X, const bf16* __restrict__ Wg,
               const bf16* __restrict__ Wu, bf16* __restrict__ H) {
  __shared__ __align__(16) bf16 sA[2][BM * LDS_STRIDE];
  __shared__ __align__(16) bf16 sG[2][BN * LDS_STRIDE];
  __shared__ __align__(16) bf16 sU[2][BN * LDS_STRIDE];

  const int tid  = threadIdx.x;
  const int lane = tid & 31;
  const int wave = tid >> 5;
  const int wm = wave >> 2;                 // 0..1 -> 64 rows
  const int wn = wave & 3;                  // 0..3 -> 32 cols
  const int m0 = blockIdx.y * BM;
  const int n0 = blockIdx.x * BN;

  // per-thread DMA geometry: two 16B chunks per tile per stage
  unsigned ldA[2], ldG[2], ldU[2], gOff[2];
  #pragma unroll
  for (int i = 0; i < 2; ++i) {
    int c = tid + i * 256, row = c >> 2, col = (c & 3) << 3;
    ldA[i] = lds_off(&sA[0][row * LDS_STRIDE + col]);
    ldG[i] = lds_off(&sG[0][row * LDS_STRIDE + col]);
    ldU[i] = lds_off(&sU[0][row * LDS_STRIDE + col]);
    gOff[i] = (unsigned)((row * D_MODEL + col) * 2);
  }

  auto issue_stage = [&](int b, int k0) {
    const bf16* pa = X  + (size_t)m0 * D_MODEL + k0;
    const bf16* pg = Wg + (size_t)n0 * D_MODEL + k0;
    const bf16* pu = Wu + (size_t)n0 * D_MODEL + k0;
    unsigned bo = (unsigned)b * TILE_BYTES;
    #pragma unroll
    for (int i = 0; i < 2; ++i) {
      async_ld_b128(ldA[i] + bo, gOff[i], pa);
      async_ld_b128(ldG[i] + bo, gOff[i], pg);
      async_ld_b128(ldU[i] + bo, gOff[i], pu);
    }
  };

  v8f zero = {};
  v8f accG[4][2], accU[4][2];
  #pragma unroll
  for (int i = 0; i < 4; ++i)
    #pragma unroll
    for (int j = 0; j < 2; ++j) { accG[i][j] = zero; accU[i][j] = zero; }

  issue_stage(0, 0);
  wait_async0();
  __syncthreads();

  const int nStages = D_MODEL / BK;         // 128
  for (int kt = 0; kt < nStages; ++kt) {
    const int cur = kt & 1;
    if (kt + 1 < nStages) issue_stage(cur ^ 1, (kt + 1) * BK);  // DMA next buffer

    v16bf af[4], gf[2], uf[2];
    #pragma unroll
    for (int mi = 0; mi < 4; ++mi)
      af[mi] = load_frag_a(&sA[cur][0], wm * 64 + mi * 16 + (lane & 15), lane);
    #pragma unroll
    for (int nj = 0; nj < 2; ++nj) {
      gf[nj] = load_frag_b(&sG[cur][0], wn * 32 + nj * 16 + (lane & 15), lane);
      uf[nj] = load_frag_b(&sU[cur][0], wn * 32 + nj * 16 + (lane & 15), lane);
    }
    #pragma unroll
    for (int mi = 0; mi < 4; ++mi)
      #pragma unroll
      for (int nj = 0; nj < 2; ++nj) {      // 16 v_wmma per wave per stage
        accG[mi][nj] = wmma_bf16(af[mi], gf[nj], accG[mi][nj]);
        accU[mi][nj] = wmma_bf16(af[mi], uf[nj], accU[mi][nj]);
      }
    wait_async0();                          // own DMA to nxt landed in LDS
    __syncthreads();                        // everyone's DMA visible
  }

  // epilogue: h = silu(g) * u  (C layout: VGPR e -> M = e + 8*(lane>=16), N = lane%16)
  #pragma unroll
  for (int mi = 0; mi < 4; ++mi)
    #pragma unroll
    for (int nj = 0; nj < 2; ++nj) {
      int col   = n0 + wn * 32 + nj * 16 + (lane & 15);
      int rbase = m0 + wm * 64 + mi * 16 + ((lane >> 4) << 3);
      #pragma unroll
      for (int e = 0; e < 8; ++e) {
        float g = accG[mi][nj][e];
        float u = accU[mi][nj][e];
        float s = g / (1.0f + __expf(-g));
        H[(size_t)(rbase + e) * D_FF + col] = f2bf(s * u);
      }
    }
}

// ---------------------------------------------------------------------------
// Kernel 2: Y = H Wd^T   (fp32 out).  grid (D_MODEL/BN=32, N_TOK/BM=32)
// ---------------------------------------------------------------------------
__global__ void __launch_bounds__(256)
down_kernel(const bf16* __restrict__ Hm, const bf16* __restrict__ Wd,
            float* __restrict__ Y) {
  __shared__ __align__(16) bf16 sA[2][BM * LDS_STRIDE];
  __shared__ __align__(16) bf16 sB[2][BN * LDS_STRIDE];

  const int tid  = threadIdx.x;
  const int lane = tid & 31;
  const int wave = tid >> 5;
  const int wm = wave >> 2;
  const int wn = wave & 3;
  const int m0 = blockIdx.y * BM;
  const int n0 = blockIdx.x * BN;

  unsigned ldA[2], ldB[2], gOff[2];
  #pragma unroll
  for (int i = 0; i < 2; ++i) {
    int c = tid + i * 256, row = c >> 2, col = (c & 3) << 3;
    ldA[i] = lds_off(&sA[0][row * LDS_STRIDE + col]);
    ldB[i] = lds_off(&sB[0][row * LDS_STRIDE + col]);
    gOff[i] = (unsigned)((row * D_FF + col) * 2);
  }

  auto issue_stage = [&](int b, int k0) {
    const bf16* pa = Hm + (size_t)m0 * D_FF + k0;
    const bf16* pb = Wd + (size_t)n0 * D_FF + k0;
    unsigned bo = (unsigned)b * TILE_BYTES;
    #pragma unroll
    for (int i = 0; i < 2; ++i) {
      async_ld_b128(ldA[i] + bo, gOff[i], pa);
      async_ld_b128(ldB[i] + bo, gOff[i], pb);
    }
  };

  v8f zero = {};
  v8f acc[4][2];
  #pragma unroll
  for (int i = 0; i < 4; ++i)
    #pragma unroll
    for (int j = 0; j < 2; ++j) acc[i][j] = zero;

  issue_stage(0, 0);
  wait_async0();
  __syncthreads();

  const int nStages = D_FF / BK;            // 344
  for (int kt = 0; kt < nStages; ++kt) {
    const int cur = kt & 1;
    if (kt + 1 < nStages) issue_stage(cur ^ 1, (kt + 1) * BK);

    v16bf af[4], bfr[2];
    #pragma unroll
    for (int mi = 0; mi < 4; ++mi)
      af[mi] = load_frag_a(&sA[cur][0], wm * 64 + mi * 16 + (lane & 15), lane);
    #pragma unroll
    for (int nj = 0; nj < 2; ++nj)
      bfr[nj] = load_frag_b(&sB[cur][0], wn * 32 + nj * 16 + (lane & 15), lane);
    #pragma unroll
    for (int mi = 0; mi < 4; ++mi)
      #pragma unroll
      for (int nj = 0; nj < 2; ++nj)
        acc[mi][nj] = wmma_bf16(af[mi], bfr[nj], acc[mi][nj]);
    wait_async0();
    __syncthreads();
  }

  #pragma unroll
  for (int mi = 0; mi < 4; ++mi)
    #pragma unroll
    for (int nj = 0; nj < 2; ++nj) {
      int col   = n0 + wn * 32 + nj * 16 + (lane & 15);
      int rbase = m0 + wm * 64 + mi * 16 + ((lane >> 4) << 3);
      #pragma unroll
      for (int e = 0; e < 8; ++e)
        Y[(size_t)(rbase + e) * D_MODEL + col] = acc[mi][nj][e];
    }
}

// ---------------------------------------------------------------------------
extern "C" void kernel_launch(void* const* d_in, const int* in_sizes, int n_in,
                              void* d_out, int out_size, void* d_ws, size_t ws_size,
                              hipStream_t stream) {
  (void)in_sizes; (void)n_in; (void)out_size; (void)ws_size;
  const float* x  = (const float*)d_in[0];
  const float* Wg = (const float*)d_in[1];
  const float* Wu = (const float*)d_in[2];
  const float* Wd = (const float*)d_in[3];
  float* y = (float*)d_out;

  // workspace layout (bf16): x | Wg | Wu | Wd | h   (~394 MB total)
  bf16* xb  = (bf16*)d_ws;
  bf16* wgb = xb  + (size_t)N_TOK * D_MODEL;
  bf16* wub = wgb + (size_t)D_FF  * D_MODEL;
  bf16* wdb = wub + (size_t)D_FF  * D_MODEL;
  bf16* hb  = wdb + (size_t)D_MODEL * D_FF;

  auto launch_cvt = [&](const float* in, bf16* out, size_t n) {
    int n8 = (int)(n / 8);
    int blocks = (n8 + 255) / 256;
    if (blocks > 8192) blocks = 8192;
    cvt_f32_bf16<<<dim3(blocks), dim3(256), 0, stream>>>(in, out, n8);
  };
  launch_cvt(x,  xb,  (size_t)N_TOK * D_MODEL);
  launch_cvt(Wg, wgb, (size_t)D_FF  * D_MODEL);
  launch_cvt(Wu, wub, (size_t)D_FF  * D_MODEL);
  launch_cvt(Wd, wdb, (size_t)D_MODEL * D_FF);

  gate_up_kernel<<<dim3(D_FF / BN, N_TOK / BM), dim3(256), 0, stream>>>(xb, wgb, wub, hb);
  down_kernel<<<dim3(D_MODEL / BN, N_TOK / BM), dim3(256), 0, stream>>>(hb, wdb, y);
}
